// NLPModel_12670153523892
// MI455X (gfx1250) — matrix operand order repaired and checked by reading, not compiled, not run on hardware
//
#include <hip/hip_runtime.h>
#include <hip/hip_bf16.h>

typedef __attribute__((ext_vector_type(16))) _Float16 v16h;
typedef __attribute__((ext_vector_type(8)))  _Float16 v8h;
typedef __attribute__((ext_vector_type(8)))  float    v8f;

#define BB    2
#define SS    2048
#define MM    (BB*SS)        // 4096 rows
#define EMB   768
#define HID   1024
#define VOCAB 50257
#define NFC   50304          // VOCAB padded to multiple of 64

// ---------------------------------------------------------------------------
// Embedding gather: x0[row,:] = emb_W[ids[row],:]
// ---------------------------------------------------------------------------
__global__ void embed_kernel(const int* __restrict__ ids,
                             const float* __restrict__ W,
                             float* __restrict__ X, int D)
{
    int row = blockIdx.x;
    int id  = ids[row];
    const float* src = W + (size_t)id * D;
    float* dst = X + (size_t)row * D;
    for (int i = threadIdx.x; i < D; i += blockDim.x) dst[i] = src[i];
}

// ---------------------------------------------------------------------------
// Weight fp32 [N,K] -> f16 [Npad,K], zero-padded rows beyond N
// ---------------------------------------------------------------------------
__global__ void w_to_f16(const float* __restrict__ W, _Float16* __restrict__ Wh,
                         int N, int K, long long total)
{
    long long idx = (long long)blockIdx.x * blockDim.x + threadIdx.x;
    if (idx >= total) return;
    long long n = idx / K;
    long long k = idx - n * K;
    Wh[idx] = (n < N) ? (_Float16)W[n * (long long)K + k] : (_Float16)0.0f;
}

// fp32 -> f16 elementwise
__global__ void x_to_f16(const float* __restrict__ X, _Float16* __restrict__ Y,
                         long long n)
{
    long long idx = (long long)blockIdx.x * blockDim.x + threadIdx.x;
    if (idx < n) Y[idx] = (_Float16)X[idx];
}

// ---------------------------------------------------------------------------
// LayerNorm a row of length D (population variance), output f16
// ---------------------------------------------------------------------------
__launch_bounds__(256)
__global__ void ln_to_f16(const float* __restrict__ X, const float* __restrict__ w,
                          const float* __restrict__ b, _Float16* __restrict__ Y,
                          int D, float eps)
{
    __shared__ float s1[256];
    __shared__ float s2[256];
    int row = blockIdx.x;
    const float* x = X + (size_t)row * D;
    float sum = 0.f, sq = 0.f;
    for (int i = threadIdx.x; i < D; i += 256) { float v = x[i]; sum += v; sq += v * v; }
    s1[threadIdx.x] = sum; s2[threadIdx.x] = sq;
    __syncthreads();
    for (int off = 128; off > 0; off >>= 1) {
        if (threadIdx.x < off) {
            s1[threadIdx.x] += s1[threadIdx.x + off];
            s2[threadIdx.x] += s2[threadIdx.x + off];
        }
        __syncthreads();
    }
    float mu   = s1[0] / (float)D;
    float var  = s2[0] / (float)D - mu * mu;
    float rstd = rsqrtf(var + eps);
    _Float16* y = Y + (size_t)row * D;
    for (int i = threadIdx.x; i < D; i += 256)
        y[i] = (_Float16)((x[i] - mu) * rstd * w[i] + b[i]);
}

// ---------------------------------------------------------------------------
// GEMM: C[M,ldc] = A[M,K](f16) * W^T   (W stored [Nb,K] f16, row-major)
// Block (8 waves) computes 128M x 64N. The 64-column B panel is staged in LDS
// in 64-K-half stages, double-buffered with GLOBAL_LOAD_ASYNC_TO_LDS_B128
// (ASYNCcnt) and shared by all 8 waves; each wave runs 8x
// V_WMMA_F32_16X16X32_F16 per stage. A/B fragments use contiguous k order
// (lane half h holds k = 16h..16h+15); D uses the WMMA layout.
// ---------------------------------------------------------------------------
__launch_bounds__(256)
__global__ void gemm_wmma_f16(const _Float16* __restrict__ A,
                              const _Float16* __restrict__ B,
                              float* __restrict__ C,
                              const float* __restrict__ bias,
                              int M, int K, int Nstore, int ldc, int Nb)
{
    // 2 buffers x (64 cols x 64 halves) = 2 x 8KB
    __shared__ __align__(32) _Float16 ldsB[2][64 * 64];

    const int tid  = threadIdx.x;
    const int lane = tid & 31;
    const int wave = tid >> 5;
    const int ngroups = Nb >> 6;                 // 64-column groups
    const int bm = blockIdx.x / ngroups;         // 128-row block index
    const int tn = blockIdx.x - bm * ngroups;

    const int row  = lane & 15;
    const int half = lane >> 4;
    const int m0   = bm * 128 + wave * 16;       // this wave's 16 rows
    const size_t n0   = (size_t)tn * 64;
    const size_t aoff = (size_t)(m0 + row) * K;

    // async-copy mapping: 64 cols x 128B per stage; thread moves 2 x 16B
    const int ccol = tid >> 2;                   // 0..63
    const int csub = tid & 3;                    // 0..3
    const _Float16* gsrc = B + (n0 + ccol) * (size_t)K;
    const uint32_t ldsc = (uint32_t)(uintptr_t)(&ldsB[0][0]) +
                          (uint32_t)(ccol * 128);        // byte addr of this col

    auto stage = [&](int st) {                   // async-copy stage 'st'
        uint64_t g0 = (uint64_t)(uintptr_t)(gsrc + (st << 6) + csub * 8);
        uint64_t g1 = (uint64_t)(uintptr_t)(gsrc + (st << 6) + csub * 8 + 32);
        uint32_t l  = ldsc + (uint32_t)((st & 1) * 8192);
        uint32_t l0 = l + (uint32_t)(csub * 16);
        uint32_t l1 = l + (uint32_t)(csub * 16 + 64);
        asm volatile("global_load_async_to_lds_b128 %0, %1, off"
                     :: "v"(l0), "v"(g0) : "memory");
        asm volatile("global_load_async_to_lds_b128 %0, %1, off"
                     :: "v"(l1), "v"(g1) : "memory");
    };

    v8f acc[4] = {};

    auto compute = [&](int st) {                 // consume staged buffer 'st'
        const _Float16* bbuf = &ldsB[st & 1][0];
#pragma unroll
        for (int kk = 0; kk < 2; ++kk) {
            const int k0 = (st << 6) + kk * 32;
            // A fragment (16x32): lanes 0-15 K{0..15}; 16-31 K{16..31}
            union { v16h v; v8h h[2]; } au;
            au.h[0] = *(const v8h*)(A + aoff + k0 + half * 16);
            au.h[1] = *(const v8h*)(A + aoff + k0 + half * 16 + 8);
            // load all four B fragments, then issue WMMAs back-to-back
            const _Float16* bk = bbuf + kk * 32 + half * 16;
            v16h bv0 = *(const v16h*)(bk + (0 * 16 + row) * 64);
            v16h bv1 = *(const v16h*)(bk + (1 * 16 + row) * 64);
            v16h bv2 = *(const v16h*)(bk + (2 * 16 + row) * 64);
            v16h bv3 = *(const v16h*)(bk + (3 * 16 + row) * 64);
            acc[0] = __builtin_amdgcn_wmma_f32_16x16x32_f16(
                false, au.v, false, bv0, (short)0, acc[0], false, false);
            acc[1] = __builtin_amdgcn_wmma_f32_16x16x32_f16(
                false, au.v, false, bv1, (short)0, acc[1], false, false);
            acc[2] = __builtin_amdgcn_wmma_f32_16x16x32_f16(
                false, au.v, false, bv2, (short)0, acc[2], false, false);
            acc[3] = __builtin_amdgcn_wmma_f32_16x16x32_f16(
                false, au.v, false, bv3, (short)0, acc[3], false, false);
        }
    };

    const int nsteps = K >> 6;                   // 64-half stages
    stage(0);                                    // prologue

    for (int st = 0; st < nsteps - 1; ++st) {    // steady state: branch-free
        stage(st + 1);
        asm volatile("s_wait_asynccnt 0x2" ::: "memory");  // stage 'st' landed
        __syncthreads();
        __builtin_prefetch(A + aoff + (st << 6) + 128, 0, 3);
        compute(st);
        __syncthreads();                         // reads done before reuse
    }
    // last stage
    asm volatile("s_wait_asynccnt 0x0" ::: "memory");
    __syncthreads();
    compute(nsteps - 1);

    // D layout: VGPR r -> row (r + 8*half), col (lane&15)
#pragma unroll
    for (int nt = 0; nt < 4; ++nt) {
        int n = (int)n0 + nt * 16 + row;
        if (n < Nstore) {
            float bb = bias ? bias[n] : 0.0f;
#pragma unroll
            for (int r = 0; r < 8; ++r) {
                int m = m0 + r + half * 8;
                C[(size_t)m * ldc + n] = acc[nt][r] + bb;
            }
        }
    }
}

// ---------------------------------------------------------------------------
// Convert gh = [gate | hidden] (width 2H) in-place to [coeff | value]:
//   coeff = 1 - sigmoid(gate),  value = sigmoid(gate) * g(hidden)
//   g(x) = x + 0.5 (x>=0) else sigmoid(x)
// ---------------------------------------------------------------------------
__global__ void gate_ab(float* __restrict__ gh, int H, long long total)
{
    long long idx = (long long)blockIdx.x * blockDim.x + threadIdx.x;
    if (idx >= total) return;
    long long r = idx / H;
    int h = (int)(idx - r * H);
    float* p = gh + r * (2LL * H);
    float g  = p[h];
    float hv = p[H + h];
    float z  = 1.0f / (1.0f + expf(-g));
    float gg = (hv >= 0.0f) ? (hv + 0.5f) : (1.0f / (1.0f + expf(-hv)));
    p[h]     = 1.0f - z;   // coeff
    p[H + h] = z * gg;     // value
}

// ---------------------------------------------------------------------------
// Sequential recurrence per (b,h) stream: h_s = coeff_s*h_{s-1} + value_s
// out = h + residual. h_{-1} = 0.5.
// ---------------------------------------------------------------------------
__global__ void scan_kernel(const float* __restrict__ gh,
                            const float* __restrict__ res,
                            float* __restrict__ out,
                            int S, int H, int nstreams)
{
    int t = blockIdx.x * blockDim.x + threadIdx.x;
    if (t >= nstreams) return;
    int b = t / H, h = t - b * H;
    const float* pa = gh  + (size_t)b * S * 2 * H + h;   // coeff at +0, value at +H
    const float* pr = res + (size_t)b * S * H + h;
    float*       po = out + (size_t)b * S * H + h;
    float hc = 0.5f;
    for (int s = 0; s < S; ++s) {
        size_t o2 = (size_t)s * 2 * H;
        float a = pa[o2];
        float v = pa[o2 + H];
        hc = fmaf(a, hc, v);
        po[(size_t)s * H] = hc + pr[(size_t)s * H];
    }
}

// ---------------------------------------------------------------------------
extern "C" void kernel_launch(void* const* d_in, const int* in_sizes, int n_in,
                              void* d_out, int out_size, void* d_ws, size_t ws_size,
                              hipStream_t stream)
{
    const int*   ids      = (const int*)  d_in[0];
    const float* emb_W    = (const float*)d_in[1];
    const float* ln0_w    = (const float*)d_in[2];
    const float* ln0_b    = (const float*)d_in[3];
    const float* gates0_W = (const float*)d_in[4];
    const float* ln1_w    = (const float*)d_in[5];
    const float* ln1_b    = (const float*)d_in[6];
    const float* gates1_W = (const float*)d_in[7];
    const float* ln2_w    = (const float*)d_in[8];
    const float* ln2_b    = (const float*)d_in[9];
    const float* gates2_W = (const float*)d_in[10];
    const float* res0_W   = (const float*)d_in[11];
    const float* lnf_w    = (const float*)d_in[12];
    const float* lnf_b    = (const float*)d_in[13];
    const float* fc_W     = (const float*)d_in[14];
    const float* fc_b     = (const float*)d_in[15];
    float* out = (float*)d_out;

    // ---- workspace layout (256B aligned segments) ----
    char* ws = (char*)d_ws;
    size_t off = 0;
    auto take = [&](size_t bytes) -> void* {
        void* p = ws + off;
        off = (off + bytes + 255) & ~(size_t)255;
        return p;
    };
    _Float16* wf_g0 = (_Float16*)take((size_t)2048 * 768  * 2);
    _Float16* wf_r0 = (_Float16*)take((size_t)1024 * 768  * 2);
    _Float16* wf_g1 = (_Float16*)take((size_t)2048 * 1024 * 2);
    _Float16* wf_g2 = (_Float16*)take((size_t)2048 * 1024 * 2);
    _Float16* wf_fc = (_Float16*)take((size_t)NFC  * 1024 * 2);
    float*    x0    = (float*)   take((size_t)MM * EMB * 4);
    _Float16* x0h   = (_Float16*)take((size_t)MM * EMB * 2);
    float*    x1    = (float*)   take((size_t)MM * HID * 4);
    float*    x2    = (float*)   take((size_t)MM * HID * 4);
    _Float16* xnh   = (_Float16*)take((size_t)MM * HID * 2);
    float*    gh    = (float*)   take((size_t)MM * 2 * HID * 4);
    float*    resb  = (float*)   take((size_t)MM * HID * 4);

    const int T = 256;
    auto nblk = [](long long n, int t) { return (int)((n + t - 1) / t); };

    // ---- convert weights to f16 ----
    { long long n = (long long)2048 * 768;
      w_to_f16<<<nblk(n, T), T, 0, stream>>>(gates0_W, wf_g0, 2048, 768, n); }
    { long long n = (long long)1024 * 768;
      w_to_f16<<<nblk(n, T), T, 0, stream>>>(res0_W,   wf_r0, 1024, 768, n); }
    { long long n = (long long)2048 * 1024;
      w_to_f16<<<nblk(n, T), T, 0, stream>>>(gates1_W, wf_g1, 2048, 1024, n); }
    { long long n = (long long)2048 * 1024;
      w_to_f16<<<nblk(n, T), T, 0, stream>>>(gates2_W, wf_g2, 2048, 1024, n); }
    { long long n = (long long)NFC * 1024;
      w_to_f16<<<nblk(n, T), T, 0, stream>>>(fc_W,     wf_fc, VOCAB, 1024, n); }

    // ---- embedding ----
    embed_kernel<<<MM, T, 0, stream>>>(ids, emb_W, x0, EMB);

    auto gemm = [&](const _Float16* A, const _Float16* B, float* C,
                    const float* bias, int M, int K, int Nstore, int ldc, int Nb) {
        int blocks = (M >> 7) * (Nb >> 6);       // 128M x 64N per block
        gemm_wmma_f16<<<blocks, 256, 0, stream>>>(A, B, C, bias,
                                                  M, K, Nstore, ldc, Nb);
    };

    // ---- layer 0: 768 -> 1024 with projected residual ----
    ln_to_f16<<<MM, 256, 0, stream>>>(x0, ln0_w, ln0_b, xnh, EMB, 1e-5f);
    x_to_f16<<<nblk((long long)MM * EMB, T), T, 0, stream>>>(x0, x0h, (long long)MM * EMB);
    gemm(xnh, wf_g0, gh,   nullptr, MM, EMB, 2048, 2048, 2048);
    gemm(x0h, wf_r0, resb, nullptr, MM, EMB, 1024, 1024, 1024);
    gate_ab<<<nblk((long long)MM * HID, T), T, 0, stream>>>(gh, HID, (long long)MM * HID);
    scan_kernel<<<nblk(BB * HID, T), T, 0, stream>>>(gh, resb, x1, SS, HID, BB * HID);

    // ---- layer 1: 1024 -> 1024, identity residual ----
    ln_to_f16<<<MM, 256, 0, stream>>>(x1, ln1_w, ln1_b, xnh, HID, 1e-5f);
    gemm(xnh, wf_g1, gh, nullptr, MM, HID, 2048, 2048, 2048);
    gate_ab<<<nblk((long long)MM * HID, T), T, 0, stream>>>(gh, HID, (long long)MM * HID);
    scan_kernel<<<nblk(BB * HID, T), T, 0, stream>>>(gh, x1, x2, SS, HID, BB * HID);

    // ---- layer 2 ----
    ln_to_f16<<<MM, 256, 0, stream>>>(x2, ln2_w, ln2_b, xnh, HID, 1e-5f);
    gemm(xnh, wf_g2, gh, nullptr, MM, HID, 2048, 2048, 2048);
    gate_ab<<<nblk((long long)MM * HID, T), T, 0, stream>>>(gh, HID, (long long)MM * HID);
    scan_kernel<<<nblk(BB * HID, T), T, 0, stream>>>(gh, x2, x1, SS, HID, BB * HID);

    // ---- final LN (eps = 0) + vocab projection with bias ----
    ln_to_f16<<<MM, 256, 0, stream>>>(x1, lnf_w, lnf_b, xnh, HID, 0.0f);
    gemm(xnh, wf_fc, out, fc_b, MM, HID, VOCAB, VOCAB, NFC);
}
